// TemporalHeteroGNN_78804059947118
// MI455X (gfx1250) — matrix operand order repaired and checked by reading, not compiled, not run on hardware
//
#include <hip/hip_runtime.h>
#include <hip/hip_bf16.h>
#include <math.h>

// Temporal hetero-GNN (HGT x2 layers, T=4, GRU, link-pred MLP) for gfx1250.
// Dense GEMMs: v_wmma_f32_16x16x32_bf16 with LDS-staged fragments:
//   coalesced global float4 loads -> v_cvt_pk_bf16 -> single ds_store_b128
//   per thread (fragment-order permuted) -> per-lane ds_load_b128 reads.
// Per-edge head matmuls are algebraically hoisted to per-node GEMMs:
//   score = k_src . (A_h q_dst)   -> qa = Q_h @ A_h^T  (dense, WMMA)
//   msg   = (v_src M_h) * alpha   -> vm = V_h @ M_h    (dense, WMMA)
// Edge passes are then pure gather/scatter served out of the 192MB L2.

namespace {

constexpr int kNC = 256, kNP = 8192, kHid = 128, kHeads = 4, kLayers = 2,
              kT = 4, kE = 250000, kB = 4096, kD = 32, kSeq = kNC + kNP;

typedef __attribute__((ext_vector_type(16))) __bf16 bf16x16;
typedef __attribute__((ext_vector_type(8)))  __bf16 bf16x8;
typedef __attribute__((ext_vector_type(8)))  float  f32x8;

// ---- WMMA wave32 fragment element -> K maps (CDNA5 ISA §7.12.2)
// A (16x32 bf16): lane = m + 16*h,  element i holds K = 16*(i/8) + 8*h + (i&7)
// B (32x16 bf16): lane = n + 16*h,  element i holds K = 16*(i/8) + 8*h + (i&7) (same as A)
__device__ __forceinline__ int akmap(int i, int h) {
  return 16 * (i >> 3) + 8 * h + (i & 7);
}

__device__ __forceinline__ float sigmoid_f(float x) { return 1.0f / (1.0f + __expf(-x)); }

__device__ __forceinline__ void atomic_max_f32(float* addr, float val) {
  unsigned int* ua = reinterpret_cast<unsigned int*>(addr);
  unsigned int cur = __float_as_uint(-3.0e38f);
  cur = atomicCAS(ua, cur, cur);  // seed with a read
  while (__uint_as_float(cur) < val) {
    unsigned int assumed = cur;
    cur = atomicCAS(ua, assumed, __float_as_uint(val));
    if (cur == assumed) break;
  }
}

// ---------------------------------------------------------------------------
// WMMA GEMM with LDS-staged fragments.
//   C[M,N] = act( A[M,K] @ B + bias ),  A f32 row-major (16B-aligned rows),
//   B element [k][n] at B[k*brs + n*bcs] (normal or transposed weights).
// Block: 128 threads = 4 waves in a 2x2 grid -> 32x32 C tile per block.
// Optional head batching via gridDim.y with per-head A/B/C strides.
// Requires: M%32==0, N%32==0, K%32==0, lda%4==0.
__global__ void k_gemm(const float* __restrict__ A, int lda, long a_hs,
                       const float* __restrict__ B, int brs, int bcs, long b_hs,
                       float* __restrict__ C, int ldc, long c_hs,
                       const float* __restrict__ bias,
                       int M, int N, int K, int act) {
  A += (long)blockIdx.y * a_hs;
  B += (long)blockIdx.y * b_hs;
  C += (long)blockIdx.y * c_hs;

  const int t = threadIdx.x;
  const int lane = t & 31;
  const int wave = t >> 5;
  const int mw = wave >> 1, nw = wave & 1;  // 2x2 wave grid
  const int nb = N >> 5;
  const int bm = blockIdx.x / nb, bn = blockIdx.x % nb;
  const int row0 = bm << 5, col0 = bn << 5;

  // LDS tiles, pre-permuted into per-lane fragment order: [subtile][lane*16+i]
  __shared__ alignas(32) __bf16 lA[2][512];
  __shared__ alignas(32) __bf16 lB[2][512];

  // A staging: thread owns (row am, 8 consecutive k at ak). Within an 8-aligned
  // k-group the fragment permutation is contiguous -> one 16B LDS store.
  const int am = t >> 2;                 // 0..31
  const int ak = (t & 3) << 3;           // 0,8,16,24
  const int ah = (ak >> 3) & 1;
  const int ai = (ak >> 4) << 3;         // 0 or 8
  __bf16* aDst = &lA[am >> 4][((am & 15) + (ah << 4)) * 16 + ai];

  // B staging: thread owns (col bn2, 8 consecutive k at bkb). Adjacent threads
  // read adjacent n -> coalesced global lines; one 16B LDS store per thread.
  const int bn2 = t & 31;                // 0..31
  const int bkb = (t >> 5) << 3;         // 0,8,16,24
  const int bh = (bkb >> 3) & 1;
  const int bi = (bkb >> 4) << 3;        // 0 or 8
  __bf16* bDst = &lB[bn2 >> 4][((bn2 & 15) + (bh << 4)) * 16 + bi];
  const float* bSrc = B + (size_t)(col0 + bn2) * bcs;

  f32x8 acc = {0.f, 0.f, 0.f, 0.f, 0.f, 0.f, 0.f, 0.f};

  for (int k0 = 0; k0 < K; k0 += 32) {
    // ---- stage A: 2 coalesced float4 loads -> packed bf16x8 -> ds_store_b128
    const float* ap = A + (size_t)(row0 + am) * lda + k0 + ak;
    float4 a0 = *(const float4*)(ap);
    float4 a1 = *(const float4*)(ap + 4);
    if (k0 + 32 < K)  // speculative prefetch of next K slab
      __builtin_prefetch(ap + 32, 0, 1);
    bf16x8 av;
    av[0] = (__bf16)a0.x; av[1] = (__bf16)a0.y;
    av[2] = (__bf16)a0.z; av[3] = (__bf16)a0.w;
    av[4] = (__bf16)a1.x; av[5] = (__bf16)a1.y;
    av[6] = (__bf16)a1.z; av[7] = (__bf16)a1.w;
    *(bf16x8*)aDst = av;

    // ---- stage B: 8 strided loads (coalesced across threads) -> one b128
    bf16x8 bv;
#pragma unroll
    for (int j = 0; j < 8; ++j)
      bv[j] = (__bf16)bSrc[(size_t)(k0 + bkb + j) * brs];
    *(bf16x8*)bDst = bv;
    __syncthreads();

    // ---- per-lane fragment reads (2x ds_load_b128 each) + WMMA ----
    bf16x16 af = *(const bf16x16*)&lA[mw][lane * 16];
    bf16x16 bf = *(const bf16x16*)&lB[nw][lane * 16];
    acc = __builtin_amdgcn_wmma_f32_16x16x32_bf16(false, af, false, bf,
                                                  (short)0, acc, false, false);
    __syncthreads();
  }

  const int n = col0 + (nw << 4) + (lane & 15);
  const float bv = bias ? bias[n] : 0.0f;
  const int half = lane >> 4;
#pragma unroll
  for (int r = 0; r < 8; ++r) {
    const int m = row0 + (mw << 4) + r + (half << 3);  // C layout: M = r + 8*half
    float v = acc[r] + bv;
    if (act == 1) v = fmaxf(v, 0.0f);
    C[(size_t)m * ldc + n] = v;
  }
}

// ---------------------------------------------------------------------------
// Link-pred layer 1: h1 = relu(concat(c_emb[tc], p_emb[tp]) @ W1 + b1)
// WMMA tiling with A rows gathered/concatenated on the fly (K=256).
__global__ void k_pair_mlp1(const float* __restrict__ cemb,
                            const float* __restrict__ pemb,
                            const int* __restrict__ tc, const int* __restrict__ tp,
                            const float* __restrict__ W1, const float* __restrict__ b1,
                            float* __restrict__ h1, int Bn) {
  const int lane = threadIdx.x & 31;
  const int wave = threadIdx.x >> 5;
  const int tn_count = kHid >> 4;                 // 8
  const int tile = blockIdx.x * (blockDim.x >> 5) + wave;
  const int tm = tile / tn_count, tn = tile % tn_count;
  const int half = lane >> 4;
  const int m = (tm << 4) + (lane & 15);
  const int n = (tn << 4) + (lane & 15);
  const int rc = tc[m] * kHid;
  const int rp = tp[m] * kHid;

  f32x8 acc = {0.f, 0.f, 0.f, 0.f, 0.f, 0.f, 0.f, 0.f};
  for (int k0 = 0; k0 < 2 * kHid; k0 += 32) {
    bf16x16 af, bf;
#pragma unroll
    for (int i = 0; i < 16; ++i) {
      const int kk = k0 + akmap(i, half);
      const float x = (kk < kHid) ? cemb[rc + kk] : pemb[rp + (kk - kHid)];
      af[i] = (__bf16)x;
    }
#pragma unroll
    for (int i = 0; i < 16; ++i)
      bf[i] = (__bf16)W1[(size_t)(k0 + akmap(i, half)) * kHid + n];
    acc = __builtin_amdgcn_wmma_f32_16x16x32_bf16(false, af, false, bf,
                                                  (short)0, acc, false, false);
  }
  const float bv = b1[n];
#pragma unroll
  for (int r = 0; r < 8; ++r) {
    const int mr = (tm << 4) + r + (half << 3);
    h1[(size_t)mr * kHid + n] = fmaxf(acc[r] + bv, 0.0f);
  }
}

// ---------------------------------------------------------------------------
// Elementwise / edge kernels
__global__ void k_fill(float* __restrict__ p, float v, int n) {
  int i = blockIdx.x * blockDim.x + threadIdx.x;
  if (i < n) p[i] = v;
}

__global__ void k_gelu(const float* __restrict__ in, float* __restrict__ out, int n) {
  int i = blockIdx.x * blockDim.x + threadIdx.x;
  if (i < n) {
    float x = in[i];
    out[i] = 0.5f * x * (1.0f + erff(x * 0.7071067811865476f));
  }
}

__global__ void k_skip_mix(const float* __restrict__ o, const float* __restrict__ skip,
                           float* __restrict__ x, int n) {
  int i = blockIdx.x * blockDim.x + threadIdx.x;
  if (i < n) {
    float a = sigmoid_f(skip[0]);
    x[i] = a * o[i] + (1.0f - a) * x[i];
  }
}

// pass 1: score[e,h] = p_rel[h]/sqrt(D) * dot32(k[src], qa[dst]); running max per (dst,h)
__global__ void k_edge_score(const int* __restrict__ src, const int* __restrict__ dst,
                             const float* __restrict__ kmat, const float* __restrict__ qa,
                             const float* __restrict__ p_rel,
                             float* __restrict__ score, float* __restrict__ mx, int E) {
  int t = blockIdx.x * blockDim.x + threadIdx.x;
  if (t >= E * kHeads) return;
  const int e = t >> 2, h = t & 3;
  const int s = src[e], d = dst[e];
  const float* kp = kmat + (size_t)s * kHid + h * kD;
  const float* qp = qa + (size_t)d * kHid + h * kD;
  float acc = 0.0f;
#pragma unroll
  for (int i = 0; i < kD; ++i) acc += kp[i] * qp[i];
  acc *= p_rel[h] * 0.1767766952966369f;  // 1/sqrt(32)
  score[t] = acc;
  atomic_max_f32(&mx[d * kHeads + h], acc);
}

// pass 2: ex = exp(score - max); sum per (dst,h)
__global__ void k_edge_exp(const int* __restrict__ dst, float* __restrict__ score,
                           const float* __restrict__ mx, float* __restrict__ sm, int E) {
  int t = blockIdx.x * blockDim.x + threadIdx.x;
  if (t >= E * kHeads) return;
  const int e = t >> 2, h = t & 3;
  const int d = dst[e];
  float ex = __expf(score[t] - mx[d * kHeads + h]);
  score[t] = ex;
  atomicAdd(&sm[d * kHeads + h], ex);
}

// pass 3: agg[dst] += alpha * vm[src]  (f32 scatter-add -> global_atomic_add_f32)
__global__ void k_edge_msg(const int* __restrict__ src, const int* __restrict__ dst,
                           const float* __restrict__ score, const float* __restrict__ sm,
                           const float* __restrict__ vmb, float* __restrict__ agg, int E) {
  int t = blockIdx.x * blockDim.x + threadIdx.x;
  if (t >= E * kHeads) return;
  const int e = t >> 2, h = t & 3;
  const int s = src[e], d = dst[e];
  const float w = score[t] / (sm[d * kHeads + h] + 1e-16f);
  const float* vp = vmb + (size_t)s * kHid + h * kD;
  float* ap = agg + (size_t)d * kHid + h * kD;
#pragma unroll
  for (int i = 0; i < kD; ++i) atomicAdd(&ap[i], w * vp[i]);
}

// GRU gates: h = (1-z)*tanh(in + r*hn) + z*h   (gi, gh already contain biases)
__global__ void k_gru_gate(const float* __restrict__ gi, const float* __restrict__ gh,
                           float* __restrict__ h, int N) {
  int t = blockIdx.x * blockDim.x + threadIdx.x;
  if (t >= N * kHid) return;
  const int n = t >> 7, j = t & (kHid - 1);
  const float* gin = gi + (size_t)n * 3 * kHid;
  const float* ghn = gh + (size_t)n * 3 * kHid;
  const float r = sigmoid_f(gin[j] + ghn[j]);
  const float z = sigmoid_f(gin[kHid + j] + ghn[kHid + j]);
  const float nn = tanhf(gin[2 * kHid + j] + r * ghn[2 * kHid + j]);
  h[t] = (1.0f - z) * nn + z * h[t];
}

// final: out = sigmoid(h1 @ W2 + b2)
__global__ void k_mlp2(const float* __restrict__ h1, const float* __restrict__ W2,
                       const float* __restrict__ b2, float* __restrict__ out, int Bn) {
  int b = blockIdx.x * blockDim.x + threadIdx.x;
  if (b >= Bn) return;
  float acc = b2[0];
  const float* r = h1 + (size_t)b * kHid;
#pragma unroll 8
  for (int i = 0; i < kHid; ++i) acc += r[i] * W2[i];
  out[b] = sigmoid_f(acc);
}

}  // namespace

// ---------------------------------------------------------------------------
extern "C" void kernel_launch(void* const* d_in, const int* in_sizes, int n_in,
                              void* d_out, int out_size, void* d_ws, size_t ws_size,
                              hipStream_t stream) {
  (void)in_sizes; (void)n_in; (void)out_size; (void)ws_size;

  // ---- input flattening (setup_inputs dict insertion order, depth-first) ----
  // 0 country_emb, 1 product_emb,
  // per layer l (base 2+24l): country{Wk,bk,Wq,bq,Wv,bv,Wa,ba,skip}=+0..+8,
  //   product{...}=+9..+17, exports{a_rel,m_rel,p_rel}=+18..+20, imports=+21..+23
  // 50 gru.Wi, 51 gru.Wh, 52 gru.bi, 53 gru.bh,
  // 54 lp.W1, 55 lp.b1, 56 lp.W2, 57 lp.b2,
  // 58 exports_edges, 59 imports_edges, 60 target_country_idx, 61 target_product_idx
  auto F = [&](int i) { return (const float*)d_in[i]; };
  auto LP = [&](int l, int k) { return (const float*)d_in[2 + 24 * l + k]; };
  const float* country_emb = F(0);
  const float* product_emb = F(1);
  const float* gWi = F(50); const float* gWh = F(51);
  const float* gbi = F(52); const float* gbh = F(53);
  const float* W1 = F(54);  const float* b1 = F(55);
  const float* W2 = F(56);  const float* b2 = F(57);
  const int* ee_all = (const int*)d_in[58];
  const int* ei_all = (const int*)d_in[59];
  const int* tci = (const int*)d_in[60];
  const int* tpi = (const int*)d_in[61];
  float* out = (float*)d_out;

  // ---- workspace layout (~92 MB of f32) ----
  float* W = (float*)d_ws;
  size_t off = 0;
  auto alloc = [&](size_t n) { float* p = W + off; off += n; return p; };
  float* xc = alloc((size_t)kNC * kHid);
  float* xp = alloc((size_t)kNP * kHid);
  float* kc = alloc((size_t)kNC * kHid);
  float* qc = alloc((size_t)kNC * kHid);
  float* vc = alloc((size_t)kNC * kHid);
  float* kp = alloc((size_t)kNP * kHid);
  float* qp = alloc((size_t)kNP * kHid);
  float* vp = alloc((size_t)kNP * kHid);
  float* qa  = alloc((size_t)kNP * kHid);   // A_h @ q per dst node (reused per relation)
  float* vmb = alloc((size_t)kNP * kHid);   // v @ M_h per src node
  float* mx  = alloc((size_t)kNP * kHeads);
  float* sm  = alloc((size_t)kNP * kHeads);
  float* alp = alloc((size_t)kE * kHeads);  // scores -> exp weights
  float* aggc = alloc((size_t)kNC * kHid);
  float* aggp = alloc((size_t)kNP * kHid);
  float* gl = alloc((size_t)kNP * kHid);
  float* ob = alloc((size_t)kNP * kHid);
  float* seq = alloc((size_t)kT * kSeq * kHid);
  float* hb  = alloc((size_t)kSeq * kHid);
  float* gi  = alloc((size_t)kSeq * 3 * kHid);
  float* gh  = alloc((size_t)kSeq * 3 * kHid);
  float* h1  = alloc((size_t)kB * kHid);

  // head-batched WMMA GEMM launcher (heads=1 for plain GEMMs)
  auto gemm_h = [&](const float* A, int lda, long a_hs,
                    const float* B, int brs, int bcs, long b_hs,
                    float* C, int ldc, long c_hs, const float* bias,
                    int M, int N, int K, int act, int heads) {
    const int blocks = (M / 32) * (N / 32);
    k_gemm<<<dim3(blocks, heads), 128, 0, stream>>>(A, lda, a_hs, B, brs, bcs, b_hs,
                                                    C, ldc, c_hs, bias, M, N, K, act);
  };
  auto gemm = [&](const float* A, int lda, const float* B, int brs, int bcs,
                  float* C, int ldc, const float* bias, int M, int N, int K, int act) {
    gemm_h(A, lda, 0, B, brs, bcs, 0, C, ldc, 0, bias, M, N, K, act, 1);
  };
  auto fill = [&](float* p, float v, int n) {
    k_fill<<<(n + 255) / 256, 256, 0, stream>>>(p, v, n);
  };
  const int egrid = (kE * kHeads + 255) / 256;

  // HGT relation: src-type (kmat,vmat,Nsrc) -> dst-type (qmat,Ndst), edges, rel params
  auto relation = [&](const float* kmat, const float* vmat, const float* qmat,
                      int Nsrc, int Ndst, const int* edges,
                      const float* a_rel, const float* m_rel, const float* p_rel,
                      float* agg) {
    // qa[n,h*32+d] = sum_f q[n,h*32+f] a[h,d,f]   (B = A_h^T ; head-batched)
    gemm_h(qmat, kHid, kD, a_rel, 1, kD, (long)kD * kD,
           qa, kHid, kD, nullptr, Ndst, kD, kD, 0, kHeads);
    // vm[n,h*32+f] = sum_d v[n,h*32+d] m[h,d,f]   (head-batched)
    gemm_h(vmat, kHid, kD, m_rel, kD, 1, (long)kD * kD,
           vmb, kHid, kD, nullptr, Nsrc, kD, kD, 0, kHeads);
    fill(mx, -3.0e38f, Ndst * kHeads);
    fill(sm, 0.0f, Ndst * kHeads);
    fill(agg, 0.0f, Ndst * kHid);
    const int* src = edges;
    const int* dst = edges + kE;
    k_edge_score<<<egrid, 256, 0, stream>>>(src, dst, kmat, qa, p_rel, alp, mx, kE);
    k_edge_exp<<<egrid, 256, 0, stream>>>(dst, alp, mx, sm, kE);
    k_edge_msg<<<egrid, 256, 0, stream>>>(src, dst, alp, sm, vmb, agg, kE);
  };

  auto update = [&](const float* agg, float* x, const float* Wa, const float* ba,
                    const float* skip, int N) {
    k_gelu<<<(N * kHid + 255) / 256, 256, 0, stream>>>(agg, gl, N * kHid);
    gemm(gl, kHid, Wa, kHid, 1, ob, kHid, ba, N, kHid, kHid, 0);
    k_skip_mix<<<(N * kHid + 255) / 256, 256, 0, stream>>>(ob, skip, x, N * kHid);
  };

  // ================= encode each snapshot =================
  for (int t = 0; t < kT; ++t) {
    hipMemcpyAsync(xc, country_emb, (size_t)kNC * kHid * 4, hipMemcpyDeviceToDevice, stream);
    hipMemcpyAsync(xp, product_emb, (size_t)kNP * kHid * 4, hipMemcpyDeviceToDevice, stream);
    const int* ee = ee_all + (size_t)t * 2 * kE;
    const int* ei = ei_all + (size_t)t * 2 * kE;
    for (int l = 0; l < kLayers; ++l) {
      // per-node-type K/Q/V linears (WMMA)
      gemm(xc, kHid, LP(l, 0), kHid, 1, kc, kHid, LP(l, 1), kNC, kHid, kHid, 0);
      gemm(xc, kHid, LP(l, 2), kHid, 1, qc, kHid, LP(l, 3), kNC, kHid, kHid, 0);
      gemm(xc, kHid, LP(l, 4), kHid, 1, vc, kHid, LP(l, 5), kNC, kHid, kHid, 0);
      gemm(xp, kHid, LP(l, 9), kHid, 1, kp, kHid, LP(l, 10), kNP, kHid, kHid, 0);
      gemm(xp, kHid, LP(l, 11), kHid, 1, qp, kHid, LP(l, 12), kNP, kHid, kHid, 0);
      gemm(xp, kHid, LP(l, 13), kHid, 1, vp, kHid, LP(l, 14), kNP, kHid, kHid, 0);
      // exports: country -> product ; imports: product -> country
      relation(kc, vc, qp, kNC, kNP, ee, LP(l, 18), LP(l, 19), LP(l, 20), aggp);
      relation(kp, vp, qc, kNP, kNC, ei, LP(l, 21), LP(l, 22), LP(l, 23), aggc);
      // gated residual updates
      update(aggc, xc, LP(l, 6), LP(l, 7), LP(l, 8), kNC);
      update(aggp, xp, LP(l, 15), LP(l, 16), LP(l, 17), kNP);
    }
    hipMemcpyAsync(seq + (size_t)t * kSeq * kHid, xc,
                   (size_t)kNC * kHid * 4, hipMemcpyDeviceToDevice, stream);
    hipMemcpyAsync(seq + (size_t)t * kSeq * kHid + (size_t)kNC * kHid, xp,
                   (size_t)kNP * kHid * 4, hipMemcpyDeviceToDevice, stream);
  }

  // ================= GRU over time (countries + products as one batch) =====
  fill(hb, 0.0f, kSeq * kHid);
  for (int t = 0; t < kT; ++t) {
    // gi = x_t @ Wi^T + bi ; gh = h @ Wh^T + bh  (B transposed: brs=1, bcs=kHid)
    gemm(seq + (size_t)t * kSeq * kHid, kHid, gWi, 1, kHid,
         gi, 3 * kHid, gbi, kSeq, 3 * kHid, kHid, 0);
    gemm(hb, kHid, gWh, 1, kHid, gh, 3 * kHid, gbh, kSeq, 3 * kHid, kHid, 0);
    k_gru_gate<<<(kSeq * kHid + 255) / 256, 256, 0, stream>>>(gi, gh, hb, kSeq);
  }

  // ================= link prediction head =================
  const float* c_emb = hb;                       // rows [0, 256)
  const float* p_emb = hb + (size_t)kNC * kHid;  // rows [256, 8448)
  {
    const int tiles = (kB / 16) * (kHid / 16);   // 2048, divisible by 4 waves/block
    k_pair_mlp1<<<tiles / 4, 128, 0, stream>>>(c_emb, p_emb, tci, tpi, W1, b1, h1, kB);
  }
  k_mlp2<<<(kB + 255) / 256, 256, 0, stream>>>(h1, W2, b2, out, kB);
}
